// TransformerBlock_51565377356453
// MI455X (gfx1250) — matrix operand order repaired and checked by reading, not compiled, not run on hardware
//
#include <hip/hip_runtime.h>
#include <hip/hip_bf16.h>
#include <math.h>

// ---------------------------------------------------------------------------
// Types for CDNA5 WMMA
// ---------------------------------------------------------------------------
typedef __bf16 bf16_t;
typedef unsigned int u32;
typedef __attribute__((ext_vector_type(16))) __bf16 v16bf;
typedef __attribute__((ext_vector_type(8)))  float  v8f;
typedef __attribute__((ext_vector_type(4)))  u32    u32x4;

union Frag16 {            // 16 bf16 elements = one WMMA A/B operand (8 VGPRs)
    v16bf v;
    u32x4 u[2];
};

static __device__ __forceinline__ v8f wmma_bf16(v16bf a, v16bf b, v8f c) {
    // D = A(16x32 bf16) * B(32x16 bf16) + C(16x16 f32)
    return __builtin_amdgcn_wmma_f32_16x16x32_bf16(
        /*neg_a=*/false, a, /*neg_b=*/false, b,
        /*c_mod=*/(short)0, c, /*reuse_a=*/false, /*reuse_b=*/false);
}

// CDNA5 async global->LDS copy (16 B per lane). Generic LDS pointer's low
// 32 bits are the LDS byte address (aperture rules, ISA 10.2).
static __device__ __forceinline__ void async_ld_b128(void* lds, const void* g) {
    asm volatile("global_load_async_to_lds_b128 %0, %1, off"
                 :: "v"((unsigned)(unsigned long long)lds), "v"(g)
                 : "memory");
}
static __device__ __forceinline__ void wait_async0() {
#if __has_builtin(__builtin_amdgcn_s_wait_asynccnt)
    __builtin_amdgcn_s_wait_asynccnt(0);
#else
    asm volatile("s_wait_asynccnt 0x0" ::: "memory");
#endif
}

// Model constants
#define DIMC     1024
#define HEADS    16
#define HEAD_DIM 64
#define SEQ      1024
#define BATCH    4
#define MROWS    (BATCH * SEQ)      // 4096
#define LN_EPS   1e-5f

// ---------------------------------------------------------------------------
// fp32 -> bf16 conversion (weights, once per launch)
// ---------------------------------------------------------------------------
__global__ void cvt_f32_bf16(const float* __restrict__ in, bf16_t* __restrict__ out, int n) {
    for (int i = blockIdx.x * blockDim.x + threadIdx.x; i < n; i += gridDim.x * blockDim.x)
        out[i] = (bf16_t)in[i];
}

// ---------------------------------------------------------------------------
// LayerNorm: one 256-thread block per row of 1024, bf16 output
// ---------------------------------------------------------------------------
__global__ __launch_bounds__(256) void ln_kernel(const float* __restrict__ x,
                                                 const float* __restrict__ g,
                                                 const float* __restrict__ b,
                                                 bf16_t* __restrict__ y) {
    __shared__ float rs[256], rq[256];
    const int row = blockIdx.x;
    const int tid = threadIdx.x;
    float xv[4];
    float s = 0.f, sq = 0.f;
#pragma unroll
    for (int j = 0; j < 4; ++j) {
        float t = x[(size_t)row * DIMC + tid + j * 256];
        xv[j] = t; s += t; sq += t * t;
    }
    rs[tid] = s; rq[tid] = sq;
    __syncthreads();
    for (int off = 128; off > 0; off >>= 1) {
        if (tid < off) { rs[tid] += rs[tid + off]; rq[tid] += rq[tid + off]; }
        __syncthreads();
    }
    const float mu  = rs[0] * (1.f / DIMC);
    const float var = rq[0] * (1.f / DIMC) - mu * mu;
    const float inv = rsqrtf(var + LN_EPS);
#pragma unroll
    for (int j = 0; j < 4; ++j) {
        int c = tid + j * 256;
        y[(size_t)row * DIMC + c] = (bf16_t)((xv[j] - mu) * inv * g[c] + b[c]);
    }
}

// ---------------------------------------------------------------------------
// Tiled bf16 WMMA GEMM:  C[M,N] = epilogue(A[M,K] @ W[K,N] + bias)
//   BM=128, BN=128, BK=32; 256 threads = 8 waves; wave owns 64x32 (4x2 tiles)
//   Tiles staged with CDNA5 async global->LDS copies (ASYNCcnt path).
// ---------------------------------------------------------------------------
#define EPI_QKV   0   // bf16 out; scale first DIMC cols (q) by 1/8
#define EPI_GELU  1   // bf16 out; exact GELU
#define EPI_RESID 2   // f32 out;  out = resid + gate[0]*val

template <int EPI>
__global__ __launch_bounds__(256) void gemm_bf16(const bf16_t* __restrict__ A,
                                                 const bf16_t* __restrict__ W,
                                                 const float*  __restrict__ bias,
                                                 const float*  __restrict__ resid,
                                                 const float*  __restrict__ gate,
                                                 float*        __restrict__ outF,
                                                 bf16_t*       __restrict__ outB,
                                                 int M, int N, int K) {
    __shared__ __align__(16) bf16_t sA[128][32 + 8];   // row stride 80 B
    __shared__ __align__(16) bf16_t sB[32][128 + 8];   // row stride 272 B

    const int tid  = threadIdx.x;
    const int lane = tid & 31;
    const int wid  = tid >> 5;
    const int mrow = lane & 15;        // M/N index inside a 16-wide tile
    const int half = lane >> 4;        // lane half selects K/M groups

    const int m0 = blockIdx.y * 128;
    const int n0 = blockIdx.x * 128;
    const int wm = (wid & 1) * 64;     // wave M offset inside block tile
    const int wn = (wid >> 1) * 32;    // wave N offset inside block tile

    v8f acc[4][2];
    v8f zero = {};
#pragma unroll
    for (int mt = 0; mt < 4; ++mt)
#pragma unroll
        for (int nt = 0; nt < 2; ++nt) acc[mt][nt] = zero;

    for (int k0 = 0; k0 < K; k0 += 32) {
        // stage A tile 128x32 (async, 16 B chunks)
#pragma unroll
        for (int i = 0; i < 2; ++i) {
            int idx = tid + i * 256;               // 0..511
            int r = idx >> 2, s = (idx & 3) * 8;
            async_ld_b128(&sA[r][s], &A[(size_t)(m0 + r) * K + k0 + s]);
        }
        // stage B tile 32x128 (async)
#pragma unroll
        for (int i = 0; i < 2; ++i) {
            int idx = tid + i * 256;
            int r = idx >> 4, s = (idx & 15) * 8;
            async_ld_b128(&sB[r][s], &W[(size_t)(k0 + r) * N + n0 + s]);
        }
        if (k0 + 32 < K) {  // CDNA5 global_prefetch_b8 of next K tiles
            __builtin_prefetch(&A[(size_t)(m0 + (tid >> 1)) * K + k0 + 32], 0, 1);
            __builtin_prefetch(&W[(size_t)(k0 + 32 + (tid >> 4)) * N + n0], 0, 1);
        }
        wait_async0();
        __syncthreads();

        Frag16 bfr[2];
#pragma unroll
        for (int nt = 0; nt < 2; ++nt) {
#pragma unroll
            for (int s = 0; s < 16; ++s)
                bfr[nt].v[s] = sB[(s & 8) * 2 + half * 8 + (s & 7)][wn + nt * 16 + mrow];
        }
#pragma unroll
        for (int mt = 0; mt < 4; ++mt) {
            Frag16 afr;
            afr.u[0] = *(const u32x4*)&sA[wm + mt * 16 + mrow][half * 8];
            afr.u[1] = *(const u32x4*)&sA[wm + mt * 16 + mrow][16 + half * 8];
#pragma unroll
            for (int nt = 0; nt < 2; ++nt)
                acc[mt][nt] = wmma_bf16(afr.v, bfr[nt].v, acc[mt][nt]);
        }
        __syncthreads();
    }

    const float gv = (EPI == EPI_RESID) ? gate[0] : 0.f;
#pragma unroll
    for (int mt = 0; mt < 4; ++mt)
#pragma unroll
        for (int nt = 0; nt < 2; ++nt)
#pragma unroll
            for (int r = 0; r < 8; ++r) {
                int gm = m0 + wm + mt * 16 + r + half * 8;
                int gn = n0 + wn + nt * 16 + mrow;
                float v = acc[mt][nt][r] + bias[gn];
                size_t o = (size_t)gm * N + gn;
                if (EPI == EPI_QKV) {
                    if (gn < DIMC) v *= 0.125f;   // q * head_dim^-0.5
                    outB[o] = (bf16_t)v;
                } else if (EPI == EPI_GELU) {
                    v = 0.5f * v * (1.f + erff(v * 0.70710678118f));
                    outB[o] = (bf16_t)v;
                } else {
                    outF[o] = resid[o] + gv * v;
                }
            }
}

// ---------------------------------------------------------------------------
// Flash-style attention. Grid: (B*HEADS, SEQ/64). Block: 256 threads (8 waves).
// qkv layout: [B*SEQ, 3*DIMC] bf16, q pre-scaled. Output bf16 [B*SEQ, DIMC].
// ---------------------------------------------------------------------------
__global__ __launch_bounds__(256) void attn_kernel(const bf16_t* __restrict__ qkv,
                                                   const int*    __restrict__ mask,
                                                   const float*  __restrict__ rel,
                                                   bf16_t*       __restrict__ out) {
    __shared__ __align__(16) bf16_t sQ[64][72];    // [q][d]      stride 144 B
    __shared__ __align__(16) bf16_t sKT[64][72];   // [d][key]    (K transposed)
    __shared__ __align__(16) bf16_t sV[64][72];    // [key][d]
    __shared__ __align__(16) bf16_t sP[64][72];    // [q][key] softmax probs
    __shared__ float sS[64][65];                   // raw scores
    __shared__ float sM[64], sL[64], sAlpha[64];

    const int tid  = threadIdx.x;
    const int lane = tid & 31;
    const int wid  = tid >> 5;
    const int mrow = lane & 15;
    const int half = lane >> 4;

    const int bh = blockIdx.x;
    const int b  = bh >> 4;
    const int h  = bh & 15;
    const int q0 = blockIdx.y * 64;

    const size_t qbase = ((size_t)(b * SEQ + q0)) * (3 * DIMC) + h * HEAD_DIM;

    // wave tile ownership: 16 output tiles (4x4), 2 per wave
    int mts[2], nts[2];
#pragma unroll
    for (int i = 0; i < 2; ++i) {
        int tt = wid * 2 + i;
        mts[i] = tt >> 2;
        nts[i] = tt & 3;
    }

    // load Q tile (64x64 bf16) via async copy; consumed after in-loop wait
#pragma unroll
    for (int i = 0; i < 2; ++i) {
        int idx = tid + i * 256;                   // 0..511
        int r = idx >> 3, s = (idx & 7) * 8;
        async_ld_b128(&sQ[r][s], &qkv[qbase + (size_t)r * (3 * DIMC) + s]);
    }
    if (tid < 64) { sM[tid] = -1e30f; sL[tid] = 0.f; }

    v8f oacc[2];
    v8f zero = {};
    oacc[0] = zero; oacc[1] = zero;
    __syncthreads();

    for (int kb = 0; kb < SEQ / 64; ++kb) {
        const int k0g = kb * 64;
        const size_t kbase = ((size_t)(b * SEQ + k0g)) * (3 * DIMC) + DIMC + h * HEAD_DIM;
        const size_t vbase = ((size_t)(b * SEQ + k0g)) * (3 * DIMC) + 2 * DIMC + h * HEAD_DIM;

        // stage K transposed: sKT[d][key] (needs per-element shuffle -> VGPR path)
        {
            int kr  = tid >> 2;                    // key row 0..63
            int ds0 = (tid & 3) * 16;              // d segment start
            union { u32x4 u[2]; bf16_t e[16]; } t;
            t.u[0] = *(const u32x4*)&qkv[kbase + (size_t)kr * (3 * DIMC) + ds0];
            t.u[1] = *(const u32x4*)&qkv[kbase + (size_t)kr * (3 * DIMC) + ds0 + 8];
#pragma unroll
            for (int j = 0; j < 16; ++j) sKT[ds0 + j][kr] = t.e[j];
        }
        // stage V: sV[key][d] (async)
#pragma unroll
        for (int i = 0; i < 2; ++i) {
            int idx = tid + i * 256;
            int r = idx >> 3, s = (idx & 7) * 8;
            async_ld_b128(&sV[r][s], &qkv[vbase + (size_t)r * (3 * DIMC) + s]);
        }
        wait_async0();        // covers V (and Q on first iteration)
        __syncthreads();

        // S = Q @ K^T  (contract over d = 64, two k-steps of 32)
        v8f sacc[2];
        sacc[0] = zero; sacc[1] = zero;
#pragma unroll
        for (int dc = 0; dc < 64; dc += 32) {
#pragma unroll
            for (int i = 0; i < 2; ++i) {
                Frag16 a, bb;
                a.u[0]  = *(const u32x4*)&sQ[mts[i] * 16 + mrow][dc + half * 8];
                a.u[1]  = *(const u32x4*)&sQ[mts[i] * 16 + mrow][dc + 16 + half * 8];
#pragma unroll
                for (int s = 0; s < 16; ++s)
                    bb.v[s] = sKT[dc + (s & 8) * 2 + half * 8 + (s & 7)][nts[i] * 16 + mrow];
                sacc[i] = wmma_bf16(a.v, bb.v, sacc[i]);
            }
        }
#pragma unroll
        for (int i = 0; i < 2; ++i)
#pragma unroll
            for (int r = 0; r < 8; ++r)
                sS[mts[i] * 16 + r + half * 8][nts[i] * 16 + mrow] = sacc[i][r];
        __syncthreads();

        // online softmax (64 rows handled by first 64 threads)
        if (tid < 64) {
            const int r = tid;
            const int q = q0 + r;
            const int mq = mask[b * SEQ + q];
            const float mold = sM[r];
            float mnew = mold;
            for (int j = 0; j < 64; ++j) {
                int key = k0g + j;
                float sv = sS[r][j] + rel[(size_t)(q - key + SEQ - 1) * HEADS + h];
                if (mq == 0 || mask[b * SEQ + key] == 0) sv = -1e9f;
                mnew = fmaxf(mnew, sv);
            }
            const float alpha = __expf(mold - mnew);
            float lsum = sL[r] * alpha;
            for (int j = 0; j < 64; ++j) {
                int key = k0g + j;
                float sv = sS[r][j] + rel[(size_t)(q - key + SEQ - 1) * HEADS + h];
                if (mq == 0 || mask[b * SEQ + key] == 0) sv = -1e9f;
                float p = __expf(sv - mnew);
                lsum += p;
                sP[r][j] = (bf16_t)p;
            }
            sM[r] = mnew; sL[r] = lsum; sAlpha[r] = alpha;
        }
        __syncthreads();

        // O = O*alpha + P @ V
#pragma unroll
        for (int i = 0; i < 2; ++i)
#pragma unroll
            for (int r = 0; r < 8; ++r)
                oacc[i][r] *= sAlpha[mts[i] * 16 + r + half * 8];
#pragma unroll
        for (int kc = 0; kc < 64; kc += 32) {
#pragma unroll
            for (int i = 0; i < 2; ++i) {
                Frag16 a, bb;
                a.u[0]  = *(const u32x4*)&sP[mts[i] * 16 + mrow][kc + half * 8];
                a.u[1]  = *(const u32x4*)&sP[mts[i] * 16 + mrow][kc + 16 + half * 8];
#pragma unroll
                for (int s = 0; s < 16; ++s)
                    bb.v[s] = sV[kc + (s & 8) * 2 + half * 8 + (s & 7)][nts[i] * 16 + mrow];
                oacc[i] = wmma_bf16(a.v, bb.v, oacc[i]);
            }
        }
        __syncthreads();
    }

    // write O / l  as bf16 into [B*SEQ, DIMC]
#pragma unroll
    for (int i = 0; i < 2; ++i)
#pragma unroll
        for (int r = 0; r < 8; ++r) {
            int row = mts[i] * 16 + r + half * 8;
            int d   = nts[i] * 16 + mrow;
            float o = oacc[i][r] / sL[row];
            out[((size_t)(b * SEQ + q0 + row)) * DIMC + h * HEAD_DIM + d] = (bf16_t)o;
        }
}

// ---------------------------------------------------------------------------
// Host-side launch
// ---------------------------------------------------------------------------
extern "C" void kernel_launch(void* const* d_in, const int* in_sizes, int n_in,
                              void* d_out, int out_size, void* d_ws, size_t ws_size,
                              hipStream_t stream) {
    const float* x      = (const float*)d_in[0];
    const int*   mask   = (const int*)  d_in[1];
    const float* ln1_g  = (const float*)d_in[2];
    const float* ln1_b  = (const float*)d_in[3];
    const float* qkv_w  = (const float*)d_in[4];
    const float* qkv_b  = (const float*)d_in[5];
    const float* proj_w = (const float*)d_in[6];
    const float* proj_b = (const float*)d_in[7];
    const float* rel    = (const float*)d_in[8];
    const float* ln2_g  = (const float*)d_in[9];
    const float* ln2_b  = (const float*)d_in[10];
    const float* mlp_w1 = (const float*)d_in[11];
    const float* mlp_b1 = (const float*)d_in[12];
    const float* mlp_w2 = (const float*)d_in[13];
    const float* mlp_b2 = (const float*)d_in[14];
    const float* gate1  = (const float*)d_in[15];
    const float* gate2  = (const float*)d_in[16];
    float* out = (float*)d_out;

    char* ws = (char*)d_ws;
    size_t off = 0;
    auto carve = [&](size_t bytes) { char* p = ws + off; off += (bytes + 255) & ~(size_t)255; return p; };

    bf16_t* wqkv  = (bf16_t*)carve((size_t)DIMC * 3 * DIMC * 2);
    bf16_t* wproj = (bf16_t*)carve((size_t)DIMC * DIMC * 2);
    bf16_t* wm1   = (bf16_t*)carve((size_t)DIMC * 4 * DIMC * 2);
    bf16_t* wm2   = (bf16_t*)carve((size_t)4 * DIMC * DIMC * 2);
    bf16_t* xn    = (bf16_t*)carve((size_t)MROWS * DIMC * 2);
    bf16_t* qkvb  = (bf16_t*)carve((size_t)MROWS * 3 * DIMC * 2);
    bf16_t* attn  = (bf16_t*)carve((size_t)MROWS * DIMC * 2);
    float*  x1    = (float*) carve((size_t)MROWS * DIMC * 4);
    bf16_t* h2    = (bf16_t*)carve((size_t)MROWS * DIMC * 2);
    bf16_t* mid   = (bf16_t*)carve((size_t)MROWS * 4 * DIMC * 2);

    // 1) weights -> bf16
    cvt_f32_bf16<<<2048, 256, 0, stream>>>(qkv_w,  wqkv,  DIMC * 3 * DIMC);
    cvt_f32_bf16<<<2048, 256, 0, stream>>>(proj_w, wproj, DIMC * DIMC);
    cvt_f32_bf16<<<2048, 256, 0, stream>>>(mlp_w1, wm1,   DIMC * 4 * DIMC);
    cvt_f32_bf16<<<2048, 256, 0, stream>>>(mlp_w2, wm2,   4 * DIMC * DIMC);

    // 2) LN1
    ln_kernel<<<MROWS, 256, 0, stream>>>(x, ln1_g, ln1_b, xn);

    // 3) QKV projection (q pre-scaled by 1/8)
    gemm_bf16<EPI_QKV><<<dim3(3 * DIMC / 128, MROWS / 128), 256, 0, stream>>>(
        xn, wqkv, qkv_b, nullptr, nullptr, nullptr, qkvb, MROWS, 3 * DIMC, DIMC);

    // 4) attention
    attn_kernel<<<dim3(BATCH * HEADS, SEQ / 64), 256, 0, stream>>>(qkvb, mask, rel, attn);

    // 5) output projection + residual:  x1 = x + gate1 * (attn @ proj_w + proj_b)
    gemm_bf16<EPI_RESID><<<dim3(DIMC / 128, MROWS / 128), 256, 0, stream>>>(
        attn, wproj, proj_b, x, gate1, x1, nullptr, MROWS, DIMC, DIMC);

    // 6) LN2
    ln_kernel<<<MROWS, 256, 0, stream>>>(x1, ln2_g, ln2_b, h2);

    // 7) MLP up + GELU
    gemm_bf16<EPI_GELU><<<dim3(4 * DIMC / 128, MROWS / 128), 256, 0, stream>>>(
        h2, wm1, mlp_b1, nullptr, nullptr, nullptr, mid, MROWS, 4 * DIMC, DIMC);

    // 8) MLP down + residual -> d_out
    gemm_bf16<EPI_RESID><<<dim3(DIMC / 128, MROWS / 128), 256, 0, stream>>>(
        mid, wm2, mlp_b2, x1, gate2, out, nullptr, MROWS, DIMC, 4 * DIMC);
}
